// SparseContextEncoder_71107478552674
// MI455X (gfx1250) — matrix operand order repaired and checked by reading, not compiled, not run on hardware
//
#include <hip/hip_runtime.h>
#include <hip/hip_bf16.h>
#include <math.h>

// ---------------------------------------------------------------------------
// SparseContextEncoder for MI455X (gfx1250, wave32, WMMA f16->f32)
// B=512, TA=6, TL=240, DIN=128, H=128, NH=4 (HD=32), NL=3, OUT=128
// ---------------------------------------------------------------------------

#define BATCH 512
#define TA 6
#define TL 240
#define HID 128

typedef _Float16 h16;
typedef __attribute__((ext_vector_type(16))) _Float16 v16h;
typedef __attribute__((ext_vector_type(8)))  float    v8f;
typedef __attribute__((ext_vector_type(4)))  float    v4f;

union Frag { v16h v; v4f q[2]; };

__device__ __forceinline__ v8f wmma_f16(v16h a, v16h b, v8f c) {
  return __builtin_amdgcn_wmma_f32_16x16x32_f16(false, a, false, b, (short)0, c,
                                                false, false);
}

// A fragment (16x32 f16) for one 16-row tile: rowptr points at this lane's row.
// lane<16: K offsets {0..15}; lane>=16: {16..31}
__device__ __forceinline__ v16h load_a_frag(const h16* rowptr, int kb, int lane) {
  const h16* p = rowptr + kb * 32 + ((lane & 16) ? 16 : 0);
  Frag f;
  f.q[0] = *(const v4f*)(p);
  f.q[1] = *(const v4f*)(p + 8);
  return f.v;
}

// B fragment (32x16 f16) from packed layout [kb][ng][lane][16 halves]
__device__ __forceinline__ v16h load_b_frag(const h16* packed, int NG, int kb,
                                            int ng, int lane) {
  const h16* p = packed + ((((size_t)kb * NG + ng) * 32 + lane) << 4);
  Frag f;
  f.q[0] = *(const v4f*)(p);
  f.q[1] = *(const v4f*)(p + 8);
  return f.v;
}

__device__ __forceinline__ float sigmf(float x) { return 1.f / (1.f + __expf(-x)); }

// ---------------------------------------------------------------------------
// Pack a (N_out, K_in) fp32 weight (used as y = x @ W^T) into f16 B-fragment
// order: B[k][n] = W[n][k], dst[((kb*(N/16)+ng)*32+lane)*16 + (k%16)]
// ---------------------------------------------------------------------------
__global__ void pack_weight_kernel(const float* __restrict__ src,
                                   h16* __restrict__ dst, int N, int K) {
  int idx = blockIdx.x * 256 + threadIdx.x;
  if (idx >= N * K) return;
  int k = idx / N, n = idx % N;
  float w = src[(size_t)n * K + k];
  int kb = k >> 5, kr = k & 31, ng = n >> 4, nl = n & 15;
  int lane = (kr < 16) ? nl : (nl + 16);
  dst[((((size_t)kb * (N >> 4) + ng) * 32 + lane) << 4) + (kr & 15)] = (h16)w;
}

__global__ void cvt_f16_kernel(const float* __restrict__ src,
                               h16* __restrict__ dst, int n) {
  int i = blockIdx.x * 256 + threadIdx.x;
  if (i < n) dst[i] = (h16)src[i];
}

// a = agent_features(B,6,4) @ W(128,4)^T + b  -> f16 (B*6,128)
__global__ void agent_proj_kernel(const float* __restrict__ af,
                                  const float* __restrict__ w,
                                  const float* __restrict__ b,
                                  h16* __restrict__ out) {
  int idx = blockIdx.x * 256 + threadIdx.x;
  if (idx >= BATCH * TA * HID) return;
  int c = idx & 127, bt = idx >> 7;
  const float* a = af + (size_t)bt * 4;
  float s = b[c];
#pragma unroll
  for (int j = 0; j < 4; ++j) s += a[j] * w[c * 4 + j];
  out[idx] = (h16)s;
}

// ---------------------------------------------------------------------------
// Generic WMMA GEMM: out(M,N) = A(M,K) f16 * Bpacked(K,N) f16 + bias
// block = 256 thr (8 waves), tile 128 rows x 128 cols, grid (M/128, N/128)
// ---------------------------------------------------------------------------
template <int K, bool OUTF32>
__global__ __launch_bounds__(256) void gemm_kernel(const h16* __restrict__ A,
                                                   const h16* __restrict__ Bp,
                                                   const float* __restrict__ bias,
                                                   void* __restrict__ out, int N) {
  constexpr int KB = K / 32;
  __shared__ __attribute__((aligned(16))) h16 sB[(size_t)K * 128];
  const int tid = threadIdx.x, lane = tid & 31, wave = tid >> 5;
  const int NG = N >> 4;
  const int cb = blockIdx.y;

  // stage this 128-col slice of packed B into LDS (layout-preserving)
  for (int i = tid; i < KB * 8 * 64; i += 256) {
    int chunk = i >> 6, r = i & 63;
    int kb = chunk >> 3, ngl = chunk & 7;
    ((v4f*)sB)[(chunk << 6) + r] =
        ((const v4f*)(Bp + (((size_t)kb * NG + (cb << 3) + ngl) << 9)))[r];
  }
  __syncthreads();

  const int row0 = blockIdx.x * 128 + wave * 16;
  const h16* arow = A + (size_t)(row0 + (lane & 15)) * K;
  v16h af[KB];
#pragma unroll
  for (int kb = 0; kb < KB; ++kb) af[kb] = load_a_frag(arow, kb, lane);

#pragma unroll
  for (int ct = 0; ct < 8; ++ct) {
    v8f acc = {};
#pragma unroll
    for (int kb = 0; kb < KB; ++kb) {
      v16h bf = load_b_frag(sB, 8, kb, ct, lane);
      acc = wmma_f16(af[kb], bf, acc);
    }
    const int ncol = (cb << 7) + (ct << 4) + (lane & 15);
    const float bv = bias ? bias[ncol] : 0.0f;
#pragma unroll
    for (int e = 0; e < 8; ++e) {
      int m = e + ((lane & 16) ? 8 : 0);
      float v = acc[e] + bv;
      size_t off = (size_t)(row0 + m) * N + ncol;
      if (OUTF32) ((float*)out)[off] = v;
      else        ((h16*)out)[off]   = (h16)v;
    }
  }
}

// ---------------------------------------------------------------------------
// Fused GRU scan (PyTorch semantics). Block = 16 batch rows, 8 waves.
// Per step each wave computes its 16 hidden columns of r/z/xn/hn via WMMA:
// r,z fuse x- and h-GEMMs; n keeps them split (n = tanh(xn + r*hn)).
// w_hh (packed f16, 96KB) lives in LDS; w_ih streamed from global (L0-hot).
// ---------------------------------------------------------------------------
__global__ __launch_bounds__(256) void gru_kernel(
    const h16* __restrict__ x, int T, const h16* __restrict__ wih,
    const h16* __restrict__ whh, const float* __restrict__ bih,
    const float* __restrict__ bhh, h16* __restrict__ y,
    float* __restrict__ sum_out) {
  __shared__ __attribute__((aligned(16))) h16 s_whh[128 * 384];   // 96 KB
  __shared__ __attribute__((aligned(16))) h16 s_h16[16 * 128];    // 4 KB
  __shared__ float s_h32[16 * 128];                               // 8 KB

  const int tid = threadIdx.x, lane = tid & 31, wave = tid >> 5;
  for (int i = tid; i < 128 * 384 / 8; i += 256)
    ((v4f*)s_whh)[i] = ((const v4f*)whh)[i];
  for (int i = tid; i < 16 * 128; i += 256) {
    s_h16[i] = (h16)0.f;
    s_h32[i] = 0.f;
  }
  __syncthreads();

  const int nl = lane & 15;
  const int ncol = (wave << 4) + nl;
  const int mbase = (lane & 16) ? 8 : 0;
  const int brow = blockIdx.x * 16;

  const float b_r = bih[ncol] + bhh[ncol];
  const float b_z = bih[128 + ncol] + bhh[128 + ncol];
  const float b_xn = bih[256 + ncol];
  const float b_hn = bhh[256 + ncol];

  const h16* xrow = x + (size_t)(brow + nl) * T * 128;
  float sacc[8] = {0, 0, 0, 0, 0, 0, 0, 0};

  for (int t = 0; t < T; ++t) {
    const h16* xp = xrow + (size_t)t * 128;
    if (t + 1 < T) __builtin_prefetch(xp + 128, 0, 1);  // global_prefetch_b8

    v16h ax[4], ah[4];
#pragma unroll
    for (int kb = 0; kb < 4; ++kb) {
      ax[kb] = load_a_frag(xp, kb, lane);
      ah[kb] = load_a_frag((const h16*)s_h16 + nl * 128, kb, lane);
    }

    v8f a_r = {}, a_z = {}, a_xn = {}, a_hn = {};
#pragma unroll
    for (int kb = 0; kb < 4; ++kb) {
      a_r  = wmma_f16(ax[kb], load_b_frag(wih,  24, kb, wave,      lane), a_r);
      a_r  = wmma_f16(ah[kb], load_b_frag(s_whh, 24, kb, wave,     lane), a_r);
      a_z  = wmma_f16(ax[kb], load_b_frag(wih,  24, kb, 8 + wave,  lane), a_z);
      a_z  = wmma_f16(ah[kb], load_b_frag(s_whh, 24, kb, 8 + wave, lane), a_z);
      a_xn = wmma_f16(ax[kb], load_b_frag(wih,  24, kb, 16 + wave, lane), a_xn);
      a_hn = wmma_f16(ah[kb], load_b_frag(s_whh, 24, kb, 16 + wave,lane), a_hn);
    }
    __syncthreads();  // everyone done reading old h

#pragma unroll
    for (int e = 0; e < 8; ++e) {
      int mm = mbase + e;
      float r = sigmf(a_r[e] + b_r);
      float z = sigmf(a_z[e] + b_z);
      float nn = tanhf(a_xn[e] + b_xn + r * (a_hn[e] + b_hn));
      float hold = s_h32[mm * 128 + ncol];
      float hnew = (1.f - z) * nn + z * hold;
      s_h32[mm * 128 + ncol] = hnew;
      s_h16[mm * 128 + ncol] = (h16)hnew;
      y[(size_t)(brow + mm) * T * 128 + (size_t)t * 128 + ncol] = (h16)hnew;
      sacc[e] += hnew;
    }
    __syncthreads();  // new h visible before next step
  }

  if (sum_out) {
#pragma unroll
    for (int e = 0; e < 8; ++e)
      sum_out[(size_t)(brow + mbase + e) * 128 + ncol] = sacc[e];
  }
}

// ---------------------------------------------------------------------------
// Self-attention core (6 tokens, 4 heads of 32): block per batch element.
// qkv (B*6,384) f16 -> o (B*6,128) f16
// ---------------------------------------------------------------------------
__global__ __launch_bounds__(256) void self_attn_core_kernel(
    const h16* __restrict__ qkv, h16* __restrict__ o) {
  const int b = blockIdx.x;
  __shared__ float sq[6 * 384];
  const h16* base = qkv + (size_t)b * 6 * 384;
  for (int i = threadIdx.x; i < 6 * 384; i += 256) sq[i] = (float)base[i];
  __syncthreads();
  const float scale = 0.1767766953f;  // 1/sqrt(32)
  for (int e = threadIdx.x; e < 6 * 128; e += 256) {
    int i = e >> 7, d = e & 127, h = d >> 5;
    float sc[6], mx = -1e30f;
    for (int j = 0; j < 6; ++j) {
      float s = 0.f;
      for (int dd = 0; dd < 32; ++dd)
        s += sq[i * 384 + h * 32 + dd] * sq[j * 384 + 128 + h * 32 + dd];
      sc[j] = s * scale;
      mx = fmaxf(mx, sc[j]);
    }
    float den = 0.f;
    for (int j = 0; j < 6; ++j) { sc[j] = __expf(sc[j] - mx); den += sc[j]; }
    float ov = 0.f;
    for (int j = 0; j < 6; ++j) ov += sc[j] * sq[j * 384 + 256 + d];
    o[(size_t)(b * 6 + i) * 128 + d] = (h16)(ov / den);
  }
}

// ---------------------------------------------------------------------------
// al cross-attention (lane queries -> 6 agent keys), mean-pooled over tokens.
// Ql (B,240,128), KV (B,6,256: K|V) -> mean_o (B,128) f16
// ---------------------------------------------------------------------------
__global__ __launch_bounds__(256) void al_core_kernel(const h16* __restrict__ Ql,
                                                      const h16* __restrict__ KV,
                                                      h16* __restrict__ mean_o) {
  const int b = blockIdx.x;
  __shared__ float s_kv[6 * 256];
  __shared__ float s_sc[8][32];
  __shared__ float s_red[8][128];
  const int tid = threadIdx.x, lane = tid & 31, wave = tid >> 5;
  const h16* kvb = KV + (size_t)b * 6 * 256;
  for (int i = tid; i < 6 * 256; i += 256) s_kv[i] = (float)kvb[i];
  __syncthreads();
  float acc[4] = {0, 0, 0, 0};
  const float scale = 0.1767766953f;
  for (int it = 0; it < TL / 8; ++it) {
    const int j = it * 8 + wave;
    const h16* q = Ql + ((size_t)b * TL + j) * 128;
    if (lane < 24) {
      int h = lane / 6, i = lane % 6;
      float s = 0.f;
      for (int dd = 0; dd < 32; ++dd)
        s += (float)q[h * 32 + dd] * s_kv[i * 256 + h * 32 + dd];
      s_sc[wave][lane] = s * scale;
    }
    __syncthreads();
#pragma unroll
    for (int rep = 0; rep < 4; ++rep) {  // head == rep, d == rep*32+lane
      float mx = -1e30f, e_[6];
      for (int i2 = 0; i2 < 6; ++i2) mx = fmaxf(mx, s_sc[wave][rep * 6 + i2]);
      float den = 0.f;
      for (int i2 = 0; i2 < 6; ++i2) {
        e_[i2] = __expf(s_sc[wave][rep * 6 + i2] - mx);
        den += e_[i2];
      }
      float ov = 0.f;
      const int d = rep * 32 + lane;
      for (int i2 = 0; i2 < 6; ++i2) ov += e_[i2] * s_kv[i2 * 256 + 128 + d];
      acc[rep] += ov / den;
    }
    __syncthreads();
  }
#pragma unroll
  for (int rep = 0; rep < 4; ++rep) s_red[wave][rep * 32 + lane] = acc[rep];
  __syncthreads();
  if (tid < 128) {
    float s = 0.f;
    for (int w = 0; w < 8; ++w) s += s_red[w][tid];
    mean_o[(size_t)b * 128 + tid] = (h16)(s * (1.0f / 240.0f));
  }
}

// ---------------------------------------------------------------------------
// la cross-attention (6 agent queries -> 240 lane keys).
// Qa (B,6,128), KV (B,240,256: K|V) -> o (B,6,128) f16
// wave owns one (query i, head h) pair per rep; lanes parallel over keys.
// ---------------------------------------------------------------------------
__global__ __launch_bounds__(256) void la_core_kernel(const h16* __restrict__ Qa,
                                                      const h16* __restrict__ KV,
                                                      h16* __restrict__ o) {
  const int b = blockIdx.x;
  const int tid = threadIdx.x, lane = tid & 31, wave = tid >> 5;
  __shared__ float s_attn[8][256];
  const float scale = 0.1767766953f;
  const h16* kvb = KV + (size_t)b * TL * 256;
  for (int rep = 0; rep < 3; ++rep) {
    const int p = rep * 8 + wave;  // 24 (i,h) pairs
    const int i = p >> 2, h = p & 3;
    float qv[32];
    const h16* qp = Qa + ((size_t)b * 6 + i) * 128 + h * 32;
#pragma unroll
    for (int dd = 0; dd < 32; ++dd) qv[dd] = (float)qp[dd];
    float sc[8], mx = -1e30f;
#pragma unroll
    for (int c = 0; c < 8; ++c) {
      const int j = c * 32 + lane;
      float s = -1e30f;
      if (j < TL) {
        s = 0.f;
        const h16* kp = kvb + (size_t)j * 256 + h * 32;
        for (int dd = 0; dd < 32; ++dd) s += qv[dd] * (float)kp[dd];
        s *= scale;
      }
      sc[c] = s;
      mx = fmaxf(mx, s);
    }
#pragma unroll
    for (int off = 16; off; off >>= 1) mx = fmaxf(mx, __shfl_xor(mx, off, 32));
    float den = 0.f;
#pragma unroll
    for (int c = 0; c < 8; ++c) {
      sc[c] = (sc[c] > -1e29f) ? __expf(sc[c] - mx) : 0.f;
      den += sc[c];
    }
#pragma unroll
    for (int off = 16; off; off >>= 1) den += __shfl_xor(den, off, 32);
    const float inv = 1.0f / den;
#pragma unroll
    for (int c = 0; c < 8; ++c) s_attn[wave][c * 32 + lane] = sc[c] * inv;
    __syncthreads();
    float ov = 0.f;
    for (int j = 0; j < TL; ++j)
      ov += s_attn[wave][j] * (float)kvb[(size_t)j * 256 + 128 + h * 32 + lane];
    o[((size_t)b * 6 + i) * 128 + h * 32 + lane] = (h16)ov;
    __syncthreads();
  }
}

// fused = [asa*sigmoid(ala_bcast) , pooled_lane*sigmoid(la)]  -> (B*6,256) f16
__global__ void fuse_kernel(const h16* __restrict__ asa,
                            const h16* __restrict__ ala,
                            const h16* __restrict__ la,
                            const float* __restrict__ lane_sum,
                            h16* __restrict__ fused) {
  int idx = blockIdx.x * 256 + threadIdx.x;
  if (idx >= BATCH * TA * 256) return;
  int c = idx & 255, bt = idx >> 8, b = bt / TA;
  float v;
  if (c < 128) {
    float a = (float)asa[(size_t)bt * 128 + c];
    float g = (float)ala[(size_t)b * 128 + c];
    v = a * sigmf(g);
  } else {
    int d = c - 128;
    float pl = lane_sum[(size_t)b * 128 + d] * (1.0f / 240.0f);
    float g = (float)la[(size_t)bt * 128 + d];
    v = pl * sigmf(g);
  }
  fused[idx] = (h16)v;
}

// ---------------------------------------------------------------------------
extern "C" void kernel_launch(void* const* d_in, const int* in_sizes, int n_in,
                              void* d_out, int out_size, void* d_ws,
                              size_t ws_size, hipStream_t stream) {
  (void)in_sizes; (void)n_in; (void)out_size; (void)ws_size;
  const float* agent_features = (const float*)d_in[0];
  const float* lane_features  = (const float*)d_in[1];
  const float* agent_proj_w   = (const float*)d_in[2];
  const float* agent_proj_b   = (const float*)d_in[3];
  const float* agru_wih = (const float*)d_in[4];
  const float* agru_whh = (const float*)d_in[5];
  const float* agru_bih = (const float*)d_in[6];
  const float* agru_bhh = (const float*)d_in[7];
  const float* lgru_wih = (const float*)d_in[8];
  const float* lgru_whh = (const float*)d_in[9];
  const float* lgru_bih = (const float*)d_in[10];
  const float* lgru_bhh = (const float*)d_in[11];
  const float* self_wqkv = (const float*)d_in[12];
  const float* self_bqkv = (const float*)d_in[13];
  const float* self_wout = (const float*)d_in[14];
  const float* self_bout = (const float*)d_in[15];
  const float* al_wqkv = (const float*)d_in[16];
  const float* al_bqkv = (const float*)d_in[17];
  const float* al_wout = (const float*)d_in[18];
  const float* al_bout = (const float*)d_in[19];
  const float* la_wqkv = (const float*)d_in[20];
  const float* la_bqkv = (const float*)d_in[21];
  const float* la_wout = (const float*)d_in[22];
  const float* la_bout = (const float*)d_in[23];
  const float* proj_w = (const float*)d_in[24];
  const float* proj_b = (const float*)d_in[25];

  size_t off = 0;
  auto alloc = [&](size_t bytes) -> void* {
    void* p = (char*)d_ws + off;
    off += (bytes + 255) & ~(size_t)255;
    return p;
  };

  // packed f16 weights
  h16* agru_wih_p = (h16*)alloc(384 * 128 * 2);
  h16* agru_whh_p = (h16*)alloc(384 * 128 * 2);
  h16* lgru_wih_p = (h16*)alloc(384 * 128 * 2);
  h16* lgru_whh_p = (h16*)alloc(384 * 128 * 2);
  h16* self_wqkv_p[3], *self_wout_p[3];
  for (int i = 0; i < 3; ++i) {
    self_wqkv_p[i] = (h16*)alloc(384 * 128 * 2);
    self_wout_p[i] = (h16*)alloc(128 * 128 * 2);
  }
  h16* al_wq_p  = (h16*)alloc(128 * 128 * 2);
  h16* al_wkv_p = (h16*)alloc(256 * 128 * 2);
  h16* al_wout_p = (h16*)alloc(128 * 128 * 2);
  h16* la_wq_p  = (h16*)alloc(128 * 128 * 2);
  h16* la_wkv_p = (h16*)alloc(256 * 128 * 2);
  h16* la_wout_p = (h16*)alloc(128 * 128 * 2);
  h16* proj_p = (h16*)alloc(256 * 128 * 2);

  // activations
  h16* lane_f16  = (h16*)alloc((size_t)BATCH * TL * HID * 2);
  h16* a_proj    = (h16*)alloc((size_t)BATCH * TA * HID * 2);
  h16* agent_enc = (h16*)alloc((size_t)BATCH * TA * HID * 2);
  h16* lane_enc  = (h16*)alloc((size_t)BATCH * TL * HID * 2);
  float* lane_sum = (float*)alloc((size_t)BATCH * HID * 4);
  h16* qkv_buf = (h16*)alloc((size_t)BATCH * TA * 384 * 2);
  h16* o_buf   = (h16*)alloc((size_t)BATCH * TA * HID * 2);
  h16* asaA    = (h16*)alloc((size_t)BATCH * TA * HID * 2);
  h16* asaB    = (h16*)alloc((size_t)BATCH * TA * HID * 2);
  h16* Q_al    = (h16*)alloc((size_t)BATCH * TL * HID * 2);
  h16* KV_al   = (h16*)alloc((size_t)BATCH * TA * 256 * 2);
  h16* mean_o  = (h16*)alloc((size_t)BATCH * HID * 2);
  h16* ala_buf = (h16*)alloc((size_t)BATCH * HID * 2);
  h16* Q_la    = (h16*)alloc((size_t)BATCH * TA * HID * 2);
  h16* KV_la   = (h16*)alloc((size_t)BATCH * TL * 256 * 2);
  h16* o_la    = (h16*)alloc((size_t)BATCH * TA * HID * 2);
  h16* la_buf  = (h16*)alloc((size_t)BATCH * TA * HID * 2);
  h16* fused   = (h16*)alloc((size_t)BATCH * TA * 256 * 2);

  auto pack = [&](const float* src, h16* dst, int N, int K) {
    pack_weight_kernel<<<(N * K + 255) / 256, 256, 0, stream>>>(src, dst, N, K);
  };
  pack(agru_wih, agru_wih_p, 384, 128);
  pack(agru_whh, agru_whh_p, 384, 128);
  pack(lgru_wih, lgru_wih_p, 384, 128);
  pack(lgru_whh, lgru_whh_p, 384, 128);
  for (int i = 0; i < 3; ++i) {
    pack(self_wqkv + (size_t)i * 384 * 128, self_wqkv_p[i], 384, 128);
    pack(self_wout + (size_t)i * 128 * 128, self_wout_p[i], 128, 128);
  }
  const float* alw = al_wqkv + (size_t)2 * 384 * 128;
  pack(alw, al_wq_p, 128, 128);
  pack(alw + 128 * 128, al_wkv_p, 256, 128);
  pack(al_wout + (size_t)2 * 128 * 128, al_wout_p, 128, 128);
  const float* law = la_wqkv + (size_t)2 * 384 * 128;
  pack(law, la_wq_p, 128, 128);
  pack(law + 128 * 128, la_wkv_p, 256, 128);
  pack(la_wout + (size_t)2 * 128 * 128, la_wout_p, 128, 128);
  pack(proj_w, proj_p, 128, 256);

  {
    int n = BATCH * TL * HID;
    cvt_f16_kernel<<<(n + 255) / 256, 256, 0, stream>>>(lane_features, lane_f16, n);
  }
  agent_proj_kernel<<<(BATCH * TA * HID + 255) / 256, 256, 0, stream>>>(
      agent_features, agent_proj_w, agent_proj_b, a_proj);

  // GRUs (batch-parallel scans)
  gru_kernel<<<BATCH / 16, 256, 0, stream>>>(a_proj, TA, agru_wih_p, agru_whh_p,
                                             agru_bih, agru_bhh, agent_enc,
                                             nullptr);
  gru_kernel<<<BATCH / 16, 256, 0, stream>>>(lane_f16, TL, lgru_wih_p,
                                             lgru_whh_p, lgru_bih, lgru_bhh,
                                             lane_enc, lane_sum);

  // 3 self-attention layers chained on asa
  const h16* asa_cur = agent_enc;
  for (int i = 0; i < 3; ++i) {
    gemm_kernel<128, false><<<dim3(BATCH * TA / 128, 3), 256, 0, stream>>>(
        asa_cur, self_wqkv_p[i], self_bqkv + (size_t)i * 384, qkv_buf, 384);
    self_attn_core_kernel<<<BATCH, 256, 0, stream>>>(qkv_buf, o_buf);
    h16* nxt = (i & 1) ? asaB : asaA;
    gemm_kernel<128, false><<<dim3(BATCH * TA / 128, 1), 256, 0, stream>>>(
        o_buf, self_wout_p[i], self_bout + (size_t)i * 128, nxt, 128);
    asa_cur = nxt;
  }

  // last-layer al cross-attention (ala/la from layers 0,1 are dead code)
  gemm_kernel<128, false><<<dim3(BATCH * TL / 128, 1), 256, 0, stream>>>(
      lane_enc, al_wq_p, al_bqkv + (size_t)2 * 384, Q_al, 128);
  gemm_kernel<128, false><<<dim3(BATCH * TA / 128, 2), 256, 0, stream>>>(
      asa_cur, al_wkv_p, al_bqkv + (size_t)2 * 384 + 128, KV_al, 256);
  al_core_kernel<<<BATCH, 256, 0, stream>>>(Q_al, KV_al, mean_o);
  gemm_kernel<128, false><<<dim3(BATCH / 128, 1), 256, 0, stream>>>(
      mean_o, al_wout_p, al_bout + (size_t)2 * 128, ala_buf, 128);

  // last-layer la cross-attention
  gemm_kernel<128, false><<<dim3(BATCH * TA / 128, 1), 256, 0, stream>>>(
      asa_cur, la_wq_p, la_bqkv + (size_t)2 * 384, Q_la, 128);
  gemm_kernel<128, false><<<dim3(BATCH * TL / 128, 2), 256, 0, stream>>>(
      lane_enc, la_wkv_p, la_bqkv + (size_t)2 * 384 + 128, KV_la, 256);
  la_core_kernel<<<BATCH, 256, 0, stream>>>(Q_la, KV_la, o_la);
  gemm_kernel<128, false><<<dim3(BATCH * TA / 128, 1), 256, 0, stream>>>(
      o_la, la_wout_p, la_bout + (size_t)2 * 128, la_buf, 128);

  // fusion + final projection (f32 out)
  fuse_kernel<<<(BATCH * TA * 256 + 255) / 256, 256, 0, stream>>>(
      asa_cur, ala_buf, la_buf, lane_sum, fused);
  gemm_kernel<256, true><<<dim3(BATCH * TA / 128, 1), 256, 0, stream>>>(
      fused, proj_p, proj_b, d_out, 128);
}
